// Aspp_Attention_12850542150043
// MI455X (gfx1250) — matrix operand order repaired and checked by reading, not compiled, not run on hardware
//
#include <hip/hip_runtime.h>
#include <hip/hip_bf16.h>
#include <math.h>

// ---------------- problem constants ----------------
#define CC   128
#define HWN  16384
#define NB   8
#define NH   8
#define HD   16
#define KV   85
#define KVP  96

typedef __attribute__((ext_vector_type(16))) __bf16 v16bf;
typedef __attribute__((ext_vector_type(8)))  __bf16 bf16x8;
typedef __attribute__((ext_vector_type(8)))  float  v8f;

union ABFrag { v16bf v; bf16x8 h[2]; };

// A-matrix fragment (16xK tile, row-major [M][stride], K-base kb):
// lane L: row = L%16, two 8-element runs at K = kb + (L/16)*8 and K = kb+16+(L/16)*8
static __device__ __forceinline__ v16bf loadA(const __bf16* base, int stride, int kb, int lane) {
  int r = lane & 15, hi = lane >> 4;
  const __bf16* p = base + r * stride + kb + hi * 8;
  ABFrag u;
  u.h[0] = *(const bf16x8*)(p);
  u.h[1] = *(const bf16x8*)(p + 16);
  return u.v;
}
// B-matrix fragment (Kx16 tile stored row-major [N][stride] i.e. transposed):
// lane L: row = L%16, two 8-element runs at K = kb + (L/16)*8 and K = kb+16+(L/16)*8
static __device__ __forceinline__ v16bf loadB(const __bf16* base, int stride, int kb, int lane) {
  int r = lane & 15, hi = lane >> 4;
  const __bf16* p = base + r * stride + kb + hi * 8;
  ABFrag u;
  u.h[0] = *(const bf16x8*)(p);
  u.h[1] = *(const bf16x8*)(p + 16);
  return u.v;
}

#define WMMA_BF16(a,b,c) __builtin_amdgcn_wmma_f32_16x16x32_bf16(false,(a),false,(b),(short)0,(c),false,false)

// ---------------- kernel 1: positional encoding table pe[c][hw] ----------------
__global__ void k_pos_enc(float* __restrict__ pe) {
  int idx = blockIdx.x * 256 + threadIdx.x;      // 128*16384 total
  int c = idx >> 14, p = idx & 16383;
  int h = p >> 7, w = p & 127;
  int axis = (c < 64) ? h : w;
  int d = c & 63;
  int i = d & 31;
  // inv[i] = 10000^(-i/32)
  float inv = __expf(-(float)i * (9.210340371976184f / 32.0f)); // ln(10000)=9.2103...
  float ang = (float)axis * inv;
  pe[idx] = (d < 32) ? sinf(ang) : cosf(ang);
}

// ---------------- kernel 2: convert Wq / Wproj to bf16 ----------------
__global__ void k_cvt_w(const float* __restrict__ Wq, const float* __restrict__ Wp,
                        __bf16* __restrict__ wqb, __bf16* __restrict__ wpb) {
  int i = blockIdx.x * 256 + threadIdx.x;        // 16384 total
  wqb[i] = (__bf16)Wq[i];
  wpb[i] = (__bf16)Wp[i];
}

// ---------------- kernel 3: 16x16 average pool of (x + pe) -> pool8[b][c][64] ----------------
__global__ void k_pool(const float* __restrict__ x, const float* __restrict__ pe,
                       float* __restrict__ pool8) {
  __shared__ float cell[64];
  int bc = blockIdx.x;                 // b*128 + c
  int c  = bc & 127;
  int tid = threadIdx.x;
  if (tid < 64) cell[tid] = 0.0f;
  __syncthreads();
  const float* xp = x  + ((size_t)bc << 14);
  const float* pp = pe + ((size_t)c  << 14);
  int col = (tid & 127) >> 4;          // pooled column for this 16-lane group
  for (int rb = 0; rb < 8; ++rb) {     // pooled row block
    float s = 0.0f;
    #pragma unroll
    for (int it = 0; it < 8; ++it) {
      int idx = (rb * 8 + it) * 256 + tid;   // two image rows per iteration, coalesced
      s += xp[idx] + pp[idx];
    }
    #pragma unroll
    for (int off = 1; off < 16; off <<= 1) s += __shfl_xor(s, off, 32);
    if ((tid & 15) == 0) atomicAdd(&cell[rb * 8 + col], s);
  }
  __syncthreads();
  if (tid < 64) pool8[(size_t)bc * 64 + tid] = cell[tid] * (1.0f / 256.0f);
}

// ---------------- kernel 4: ASPP branches + DSC + LN + GELU + KV projection ----------------
__global__ void __launch_bounds__(256) k_z(const float* __restrict__ pool8,
                                           const float* __restrict__ dw,
                                           const float* __restrict__ pw,
                                           const float* __restrict__ ln_w,
                                           const float* __restrict__ ln_b,
                                           const float* __restrict__ Wkv,
                                           __bf16* __restrict__ Kb,
                                           __bf16* __restrict__ Vt) {
  __shared__ float p8[128][64];
  __shared__ float p4[128][16];
  __shared__ float p2[128][4];
  __shared__ float p1[128];
  __shared__ float y [128][85];
  __shared__ float zb[85][128];
  int b = blockIdx.x, tid = threadIdx.x;
  const float* src = pool8 + (size_t)b * 128 * 64;
  for (int i = tid; i < 128 * 64; i += 256) (&p8[0][0])[i] = src[i];
  __syncthreads();
  for (int i = tid; i < 128 * 16; i += 256) {
    int c = i >> 4, ij = i & 15, r = ij >> 2, q = ij & 3;
    p4[c][ij] = 0.25f * (p8[c][(2*r)*8 + 2*q]     + p8[c][(2*r)*8 + 2*q + 1] +
                         p8[c][(2*r+1)*8 + 2*q]   + p8[c][(2*r+1)*8 + 2*q + 1]);
  }
  __syncthreads();
  for (int i = tid; i < 128 * 4; i += 256) {
    int c = i >> 2, ij = i & 3, r = ij >> 1, q = ij & 1;
    p2[c][ij] = 0.25f * (p4[c][(2*r)*4 + 2*q]     + p4[c][(2*r)*4 + 2*q + 1] +
                         p4[c][(2*r+1)*4 + 2*q]   + p4[c][(2*r+1)*4 + 2*q + 1]);
  }
  __syncthreads();
  if (tid < 128) p1[tid] = 0.25f * (p2[tid][0] + p2[tid][1] + p2[tid][2] + p2[tid][3]);
  __syncthreads();
  // depthwise 3x3 SAME on each branch map
  for (int t = tid; t < 128 * 85; t += 256) {
    int c = t / 85, pos = t - c * 85;
    int s, off;
    if      (pos < 64) { s = 8; off = 0;  }
    else if (pos < 80) { s = 4; off = 64; }
    else if (pos < 84) { s = 2; off = 80; }
    else               { s = 1; off = 84; }
    int lp = pos - off, i = lp / s, j = lp - (lp / s) * s;
    const float* map = (s == 8) ? &p8[c][0] : (s == 4) ? &p4[c][0]
                      : (s == 2) ? &p2[c][0] : &p1[c];
    float acc = 0.0f;
    #pragma unroll
    for (int di = 0; di < 3; ++di)
      #pragma unroll
      for (int dj = 0; dj < 3; ++dj) {
        int ii = i + di - 1, jj = j + dj - 1;
        if (ii >= 0 && ii < s && jj >= 0 && jj < s)
          acc += dw[c * 9 + di * 3 + dj] * map[ii * s + jj];
      }
    y[c][pos] = acc;
  }
  __syncthreads();
  // pointwise 1x1
  for (int t = tid; t < 85 * 128; t += 256) {
    int pos = t >> 7, co = t & 127;
    float acc = 0.0f;
    for (int ci = 0; ci < 128; ++ci) acc += pw[co * 128 + ci] * y[ci][pos];
    zb[pos][co] = acc;
  }
  __syncthreads();
  // LayerNorm + exact GELU
  if (tid < 85) {
    float mu = 0.0f;
    for (int c = 0; c < 128; ++c) mu += zb[tid][c];
    mu *= (1.0f / 128.0f);
    float var = 0.0f;
    for (int c = 0; c < 128; ++c) { float d = zb[tid][c] - mu; var += d * d; }
    var *= (1.0f / 128.0f);
    float inv = rsqrtf(var + 1e-5f);
    for (int c = 0; c < 128; ++c) {
      float v = (zb[tid][c] - mu) * inv * ln_w[c] + ln_b[c];
      zb[tid][c] = v * 0.5f * (1.0f + erff(v * 0.70710678118654752f));
    }
  }
  __syncthreads();
  // K -> [b][m][96][32] bf16 (kv padded to 96, hd padded to 32 with zeros)
  for (int t = tid; t < NH * KVP * 32; t += 256) {
    int m = t / (KVP * 32), rem = t - m * KVP * 32;
    int kvp = rem >> 5, hd = rem & 31;
    float val = 0.0f;
    if (kvp < KV && hd < 16) {
      const float* wr = Wkv + (size_t)(m * 16 + hd) * 128;
      for (int ci = 0; ci < 128; ++ci) val += wr[ci] * zb[kvp][ci];
    }
    Kb[((size_t)(b * NH + m)) * KVP * 32 + kvp * 32 + hd] = (__bf16)val;
  }
  // V^T -> [b][m][16][96] bf16
  for (int t = tid; t < NH * 16 * KVP; t += 256) {
    int m = t / (16 * KVP), rem = t - m * 16 * KVP;
    int hd = rem / KVP, kvp = rem - hd * KVP;
    float val = 0.0f;
    if (kvp < KV) {
      const float* wr = Wkv + (size_t)(128 + m * 16 + hd) * 128;
      for (int ci = 0; ci < 128; ++ci) val += wr[ci] * zb[kvp][ci];
    }
    Vt[((size_t)(b * NH + m)) * 16 * KVP + hd * KVP + kvp] = (__bf16)val;
  }
}

// ---------------- kernel 5: fused q-proj + attention + out-proj ----------------
#define XT_S 136   // 128 + 8 pad (bf16 elems)
#define QP_S 104   // 96 + 8 pad
#define SO_S 136   // 128 + 8 pad

__global__ void __launch_bounds__(256) k_attn(const float* __restrict__ x,
                                              const float* __restrict__ pe,
                                              const __bf16* __restrict__ WqB,
                                              const __bf16* __restrict__ WprojB,
                                              const __bf16* __restrict__ Kb,
                                              const __bf16* __restrict__ Vt,
                                              const float* __restrict__ bproj,
                                              float* __restrict__ out) {
  __shared__ __bf16 xt[256 * XT_S];          // x+pe tile, row-major [pos][c]
  __shared__ __bf16 stQP[8 * 16 * QP_S];     // per-wave Q/P staging (D->A relayout)
  __shared__ __bf16 stO [8 * 16 * SO_S];     // per-wave O staging
  int b = blockIdx.y;
  int tileBase = blockIdx.x * 256;
  int tid = threadIdx.x;
  // ---- phase 1: cooperative coalesced tile load + PE add + bf16 convert ----
  {
    const float* xb  = x  + ((size_t)b * CC * HWN) + tileBase;
    const float* peb = pe + tileBase;
    #pragma unroll 4
    for (int c = 0; c < 128; ++c)
      xt[tid * XT_S + c] = (__bf16)(xb[(size_t)c * HWN + tid] + peb[(size_t)c * HWN + tid]);
  }
  __syncthreads();
  int w = tid >> 5, lane = tid & 31;
  int r = lane & 15, hi = lane >> 4;
  __bf16* qp = stQP + w * 16 * QP_S;
  __bf16* so = stO  + w * 16 * SO_S;
  for (int st = 0; st < 2; ++st) {
    int rowBase = w * 32 + st * 16;
    const __bf16* aX = &xt[rowBase * XT_S];
    for (int m = 0; m < NH; ++m) {
      // ---- q tile for head m: (16 x 128) x Wq[m-tile]^T -> 16x16 ----
      v8f qd = {};
      #pragma unroll
      for (int ks = 0; ks < 4; ++ks) {
        v16bf a  = loadA(aX, XT_S, ks * 32, lane);
        v16bf bb = loadB(WqB + m * 16 * 128, 128, ks * 32, lane);
        qd = WMMA_BF16(a, bb, qd);
      }
      // stage Q as A-matrix 16x32 (cols 16..31 zero)
      #pragma unroll
      for (int e = 0; e < 8; ++e) {
        int row = e + 8 * hi;
        qp[row * QP_S + r]      = (__bf16)qd[e];
        qp[row * QP_S + 16 + r] = (__bf16)0.0f;
      }
      v16bf qa = loadA(qp, QP_S, 0, lane);
      // ---- scores: Q (16x32) x K^T (32x96) ----
      const __bf16* kbm = Kb + ((size_t)(b * NH + m)) * KVP * 32;
      v8f s[6];
      #pragma unroll
      for (int j = 0; j < 6; ++j) {
        v16bf bb = loadB(kbm + j * 16 * 32, 32, 0, lane);
        v8f zc = {};
        s[j] = WMMA_BF16(qa, bb, zc);
      }
      // ---- softmax over kv (scale 1/4, mask kv>=85) ----
      float mx[8], sm[8];
      #pragma unroll
      for (int e = 0; e < 8; ++e) {
        float mv = -3.0e38f;
        #pragma unroll
        for (int j = 0; j < 6; ++j) {
          float v = s[j][e] * 0.25f;
          if (j == 5 && r >= 5) v = -3.0e38f;   // kv = 80 + r >= 85
          s[j][e] = v;
          mv = fmaxf(mv, v);
        }
        #pragma unroll
        for (int off = 1; off < 16; off <<= 1) mv = fmaxf(mv, __shfl_xor(mv, off, 32));
        mx[e] = mv;
      }
      #pragma unroll
      for (int e = 0; e < 8; ++e) {
        float ss = 0.0f;
        #pragma unroll
        for (int j = 0; j < 6; ++j) { float pv = __expf(s[j][e] - mx[e]); s[j][e] = pv; ss += pv; }
        #pragma unroll
        for (int off = 1; off < 16; off <<= 1) ss += __shfl_xor(ss, off, 32);
        sm[e] = 1.0f / ss;
      }
      // stage P (16x96) as A-matrix
      #pragma unroll
      for (int e = 0; e < 8; ++e) {
        int row = e + 8 * hi;
        #pragma unroll
        for (int j = 0; j < 6; ++j)
          qp[row * QP_S + j * 16 + r] = (__bf16)(s[j][e] * sm[e]);
      }
      // ---- O = P (16x96) x V (96x16) ----
      const __bf16* vtm = Vt + ((size_t)(b * NH + m)) * 16 * KVP;
      v8f od = {};
      #pragma unroll
      for (int ks = 0; ks < 3; ++ks) {
        v16bf a  = loadA(qp, QP_S, ks * 32, lane);
        v16bf bb = loadB(vtm, KVP, ks * 32, lane);
        od = WMMA_BF16(a, bb, od);
      }
      #pragma unroll
      for (int e = 0; e < 8; ++e)
        so[(e + 8 * hi) * SO_S + m * 16 + r] = (__bf16)od[e];
    }
    // ---- out-proj: O (16x128) x Wproj^T + bias, scatter to (b,c,hw) ----
    for (int nt = 0; nt < 8; ++nt) {
      v8f acc = {};
      #pragma unroll
      for (int ks = 0; ks < 4; ++ks) {
        v16bf a  = loadA(so, SO_S, ks * 32, lane);
        v16bf bb = loadB(WprojB + nt * 16 * 128, 128, ks * 32, lane);
        acc = WMMA_BF16(a, bb, acc);
      }
      int c = nt * 16 + r;
      float bias = bproj[c];
      float* op = out + ((size_t)(b * CC + c) << 14) + tileBase + rowBase + 8 * hi;
      float4 o0 = make_float4(acc[0] + bias, acc[1] + bias, acc[2] + bias, acc[3] + bias);
      float4 o1 = make_float4(acc[4] + bias, acc[5] + bias, acc[6] + bias, acc[7] + bias);
      *(float4*)(op)     = o0;
      *(float4*)(op + 4) = o1;
    }
  }
}

// ---------------- host launcher ----------------
extern "C" void kernel_launch(void* const* d_in, const int* in_sizes, int n_in,
                              void* d_out, int out_size, void* d_ws, size_t ws_size,
                              hipStream_t stream) {
  const float* x     = (const float*)d_in[0];
  const float* Wq    = (const float*)d_in[1];
  const float* Wkv   = (const float*)d_in[2];
  const float* Wproj = (const float*)d_in[3];
  const float* bproj = (const float*)d_in[4];
  const float* dw_w  = (const float*)d_in[5];
  const float* pw_w  = (const float*)d_in[6];
  const float* ln_w  = (const float*)d_in[7];
  const float* ln_b  = (const float*)d_in[8];
  float* out = (float*)d_out;

  char* ws = (char*)d_ws;
  float*  pe     = (float*)(ws);                              // 8 MiB
  __bf16* WqB    = (__bf16*)(ws + 8388608);                   // 32 KiB
  __bf16* WprojB = (__bf16*)(ws + 8388608 + 32768);           // 32 KiB
  float*  pool8  = (float*)(ws + 8388608 + 65536);            // 256 KiB
  __bf16* Kb     = (__bf16*)(ws + 8388608 + 65536 + 262144);  // 384 KiB
  __bf16* Vt     = (__bf16*)(ws + 8388608 + 65536 + 262144 + 393216); // 192 KiB

  k_pos_enc<<<dim3(8192), dim3(256), 0, stream>>>(pe);
  k_cvt_w  <<<dim3(64),   dim3(256), 0, stream>>>(Wq, Wproj, WqB, WprojB);
  k_pool   <<<dim3(1024), dim3(256), 0, stream>>>(x, pe, pool8);
  k_z      <<<dim3(8),    dim3(256), 0, stream>>>(pool8, dw_w, pw_w, ln_w, ln_b, Wkv, Kb, Vt);
  k_attn   <<<dim3(64, 8), dim3(256), 0, stream>>>(x, pe, WqB, WprojB, Kb, Vt, bproj, out);
}
